// Aggregate_81192061763614
// MI455X (gfx1250) — compile-verified, matrix operand order checked by reading, not run on hardware
//
#include <hip/hip_runtime.h>
#include <cmath>

// ---------------------------------------------------------------------------
// Types for CDNA5 WMMA (wave32, 16x16x32 bf16 -> f32)
// ---------------------------------------------------------------------------
typedef __bf16 bf16_t;
typedef __attribute__((ext_vector_type(16))) __bf16 v16bf;
typedef __attribute__((ext_vector_type(8)))  __bf16 v8bf;
typedef __attribute__((ext_vector_type(8)))  float  v8f;

__device__ __host__ inline bf16_t to_bf16(float f) {
    unsigned u = __builtin_bit_cast(unsigned, f);
    u += 0x7FFFu + ((u >> 16) & 1u);              // round-to-nearest-even
    unsigned short s = (unsigned short)(u >> 16);
    return __builtin_bit_cast(bf16_t, s);
}

__device__ inline float silu_f(float x) { return x / (1.0f + __expf(-x)); }

// Load one 16x32 bf16 operand fragment (A-layout == B^T-layout, K-striped):
// lanes 0-15 hold K [0..7] and [16..23] of row (lane), lanes 16-31 hold
// K [8..15] and [24..31] of row (lane-16).  p already includes row*Kd +
// halfsel*8; chunks are 16B each -> two dwordx4 loads, unconditional
// (invalid lanes are pointed at a zeroed guard buffer by the caller).
__device__ inline v16bf load_frag16(const bf16_t* __restrict__ p) {
    const v8bf lo = *(const v8bf*)(p);
    const v8bf hi = *(const v8bf*)(p + 16);
    v16bf f;
    #pragma unroll
    for (int i = 0; i < 8; ++i) { f[i] = lo[i]; f[i + 8] = hi[i]; }
    return f;
}

// ---------------------------------------------------------------------------
// Generic (shifted-tap) GEMM:  C[r, col0+n] = act( sum_tap sum_k
//      A[r + shift0+tap, k] * Bt[tap, n, k] + bias[n] ) * scale + Res[...]
// A: bf16 [M x Kd] (row stride Kd), rows are (b*T + t); tap shifts t with
// zero padding at batch boundaries (== conv1d same-padding along T) realised
// by redirecting out-of-range lanes to the zero-guard buffer `zg` (>= Kd
// zeroed bf16 elements), keeping all fragment loads unconditional.
// Bt: bf16 [ntaps][N][Kd]  (pre-transposed weights).
// Block = 8 waves, wave tile 64x64 (4x4 WMMA accum), block tile 256x128.
// ---------------------------------------------------------------------------
__global__ __launch_bounds__(256)
void conv_gemm_bf16(const bf16_t* __restrict__ A,
                    const bf16_t* __restrict__ Bt,
                    const float*  __restrict__ bias,
                    float*        __restrict__ C,
                    bf16_t*       __restrict__ Cbf,
                    const float*  __restrict__ Res,
                    const bf16_t* __restrict__ zg,
                    int M, int N, int Kd, int T,
                    int ntaps, int shift0,
                    int ldc, int col0, int act, float scale)
{
    const int lane    = threadIdx.x & 31;
    const int wave    = threadIdx.x >> 5;
    const int waveM   = wave >> 1;                 // 0..3
    const int waveN   = wave & 1;                  // 0..1
    const int wm0     = blockIdx.x * 256 + waveM * 64;
    const int wn0     = blockIdx.y * 128 + waveN * 64;
    const int halfsel = lane >> 4;                 // K-half select
    const int lr      = lane & 15;                 // row-in-tile

    v8f acc[4][4];
    #pragma unroll
    for (int i = 0; i < 4; ++i)
        #pragma unroll
        for (int j = 0; j < 4; ++j)
            #pragma unroll
            for (int e = 0; e < 8; ++e) acc[i][j][e] = 0.0f;

    const bf16_t* zgp = zg + halfsel * 8;

    for (int tap = 0; tap < ntaps; ++tap) {
        const int shift = shift0 + tap;
        const bf16_t* Btap = Bt + (size_t)tap * (size_t)N * (size_t)Kd;

        const bf16_t* pA[4];
        #pragma unroll
        for (int i = 0; i < 4; ++i) {
            const int r  = wm0 + 16 * i + lr;
            const int t  = r % T;
            const int tt = t + shift;
            const bool ok = (r < M) && (tt >= 0) && (tt < T);
            pA[i] = ok ? (A + (size_t)(r + shift) * Kd + halfsel * 8) : zgp;
        }
        const bf16_t* pB[4];
        #pragma unroll
        for (int j = 0; j < 4; ++j) {
            const int n = wn0 + 16 * j + lr;
            pB[j] = (n < N) ? (Btap + (size_t)n * Kd + halfsel * 8) : zgp;
        }

        for (int k0 = 0; k0 < Kd; k0 += 32) {
            __builtin_prefetch(pA[0] + k0 + 32, 0, 1);   // global_prefetch_b8
            __builtin_prefetch(pB[0] + k0 + 32, 0, 1);
            v16bf af[4], bfm[4];
            #pragma unroll
            for (int i = 0; i < 4; ++i) af[i]  = load_frag16(pA[i] + k0);
            #pragma unroll
            for (int j = 0; j < 4; ++j) bfm[j] = load_frag16(pB[j] + k0);

            #pragma unroll
            for (int i = 0; i < 4; ++i)
                #pragma unroll
                for (int j = 0; j < 4; ++j)
                    acc[i][j] = __builtin_amdgcn_wmma_f32_16x16x32_bf16(
                        false, af[i], false, bfm[j], (short)0, acc[i][j],
                        false, false);
        }
    }

    // Epilogue.  C/D layout: lanes 0-15 -> N=lane, M=e; lanes 16-31 -> M=e+8.
    #pragma unroll
    for (int j = 0; j < 4; ++j) {
        const int col = wn0 + 16 * j + lr;
        if (col >= N) continue;
        const float bv = bias ? bias[col] : 0.0f;
        #pragma unroll
        for (int i = 0; i < 4; ++i) {
            #pragma unroll
            for (int e = 0; e < 8; ++e) {
                const int row = wm0 + 16 * i + e + halfsel * 8;
                if (row >= M) continue;
                float v = acc[i][j][e] + bv;
                if (act) v = fmaxf(v, 0.0f);
                v *= scale;
                const size_t idx = (size_t)row * ldc + col0 + col;
                if (Res) v += Res[idx];
                if (C)   C[idx]   = v;
                if (Cbf) Cbf[idx] = to_bf16(v);
            }
        }
    }
}

// ---------------------------------------------------------------------------
// prep kernels: zero guard, fp32 -> bf16 conversion, weight transposes
// ---------------------------------------------------------------------------
__global__ void zero_bf16(bf16_t* __restrict__ p, int n) {
    int i = blockIdx.x * blockDim.x + threadIdx.x;
    if (i < n) p[i] = __builtin_bit_cast(bf16_t, (unsigned short)0);
}

__global__ void cvt_f32_to_bf16(const float* __restrict__ src,
                                bf16_t* __restrict__ dst, int n) {
    int i = blockIdx.x * blockDim.x + threadIdx.x;
    if (i < n) dst[i] = to_bf16(src[i]);
}

// W [Kd][N] fp32 -> Wt [N][Kd] bf16
__global__ void transpose_w(const float* __restrict__ w,
                            bf16_t* __restrict__ wt, int Kd, int N) {
    int i = blockIdx.x * blockDim.x + threadIdx.x;
    if (i < Kd * N) {
        int k = i / N, n = i % N;
        wt[(size_t)n * Kd + k] = to_bf16(w[i]);
    }
}

// conv weight [O][I][K] fp32 -> wt [K][O][I] bf16 (per-tap transposed GEMM B)
__global__ void prep_conv_w(const float* __restrict__ w,
                            bf16_t* __restrict__ wt, int O, int I, int K) {
    int i = blockIdx.x * blockDim.x + threadIdx.x;
    if (i < O * I * K) {
        int k  = i % K;
        int ii = (i / K) % I;
        int o  = i / (K * I);
        wt[((size_t)k * O + o) * I + ii] = to_bf16(w[i]);
    }
}

// ---------------------------------------------------------------------------
// Depthwise causal conv (K=4) + SiLU.  xs = cols [0,D) of xz (ld = 2D).
// ---------------------------------------------------------------------------
__global__ __launch_bounds__(256)
void dwconv_silu(const float* __restrict__ xz, const float* __restrict__ w,
                 const float* __restrict__ b, float* __restrict__ u,
                 bf16_t* __restrict__ ubf, int T, int D, int ld) {
    const int d = (blockIdx.x & 3) * 256 + threadIdx.x;   // D == 1024
    const int r = blockIdx.x >> 2;
    const int t = r % T;
    const int base = r - t;                               // b*T
    float s = b[d];
    #pragma unroll
    for (int k = 0; k < 4; ++k) {
        const int tt = t - 3 + k;
        if (tt >= 0) s += w[d * 4 + k] * xz[(size_t)(base + tt) * ld + d];
    }
    const float uu = silu_f(s);
    u  [(size_t)r * D + d] = uu;
    ubf[(size_t)r * D + d] = to_bf16(uu);
}

// ---------------------------------------------------------------------------
// Selective scan: one wave32 per batch, lane = state index s (S == 32).
// h_t = h_{t-1} @ A^T + x_t * Bv ;  c_t = h_t . Cv
// ---------------------------------------------------------------------------
__global__ __launch_bounds__(32)
void scan_kernel(const float* __restrict__ xssm, const float* __restrict__ Amat,
                 const float* __restrict__ Bv, const float* __restrict__ Cv,
                 float* __restrict__ c, int T) {
    const int b    = blockIdx.x;
    const int lane = threadIdx.x;
    float arow[32];
    #pragma unroll
    for (int j = 0; j < 32; ++j) arow[j] = Amat[lane * 32 + j];
    const float bv = Bv[lane], cv = Cv[lane];
    float h = 0.0f;
    const float* xp = xssm + (size_t)b * T * 32;
    float* cp = c + (size_t)b * T;
    for (int t = 0; t < T; ++t) {
        const float xt = xp[t * 32 + lane];
        float hn = xt * bv;
        #pragma unroll
        for (int j = 0; j < 32; ++j) hn += arow[j] * __shfl(h, j, 32);
        h = hn;
        float pc = h * cv;
        #pragma unroll
        for (int off = 16; off > 0; off >>= 1) pc += __shfl_xor(pc, off, 32);
        if (lane == 0) cp[t] = pc;
    }
}

// ---------------------------------------------------------------------------
// y = (c + Dv) * u * silu(gate)   (gate = cols [D,2D) of xz), emit bf16
// ---------------------------------------------------------------------------
__global__ __launch_bounds__(256)
void mamba_y(const float* __restrict__ cbuf, const float* __restrict__ Dv,
             const float* __restrict__ u, const float* __restrict__ xz,
             bf16_t* __restrict__ ybf, int T, int D, int ld) {
    const int d = (blockIdx.x & 3) * 256 + threadIdx.x;
    const int r = blockIdx.x >> 2;
    const float g   = xz[(size_t)r * ld + D + d];
    const float val = (cbuf[r] + Dv[d]) * u[(size_t)r * D + d] * silu_f(g);
    ybf[(size_t)r * D + d] = to_bf16(val);
}

// ---------------------------------------------------------------------------
// Anomaly head: one block per batch, one thread per memory slot (M=256).
// ---------------------------------------------------------------------------
__global__ __launch_bounds__(256)
void anomaly_kernel(const float* __restrict__ out, const float* __restrict__ mem_p,
                    const float* __restrict__ mem_s, float* __restrict__ anom,
                    int T, int D) {
    __shared__ float sdp[256], sds[256];
    const int b = blockIdx.x, j = threadIdx.x;
    const float* f = out + ((size_t)b * T + (T - 1)) * D;
    float dp = 0.0f, ds = 0.0f;
    for (int d = 0; d < D; ++d) {
        const float fv = f[d];
        float a = mem_p[(size_t)j * D + d] - fv; dp += a * a;
        float e = mem_s[(size_t)j * D + d] - fv; ds += e * e;
    }
    dp = sqrtf(dp); ds = sqrtf(ds);
    sdp[j] = dp; sds[j] = ds;
    __syncthreads();
    for (int s = 128; s > 0; s >>= 1) {
        if (j < s) {
            sdp[j] = fmaxf(sdp[j], sdp[j + s]);
            sds[j] = fmaxf(sds[j], sds[j + s]);
        }
        __syncthreads();
    }
    const float dpmax = sdp[0], dsmax = sds[0];
    __syncthreads();
    sdp[j] = fminf(dp / (dpmax + 1e-6f), ds / (dsmax + 1e-6f));
    __syncthreads();
    for (int s = 128; s > 0; s >>= 1) {
        if (j < s) sdp[j] = fminf(sdp[j], sdp[j + s]);
        __syncthreads();
    }
    if (j == 0) anom[b] = sdp[0];
}

// ---------------------------------------------------------------------------
// Host-side: one Mamba block
// ---------------------------------------------------------------------------
static void mamba_block(hipStream_t stream,
                        const bf16_t* Xbf, const float* in_b,
                        const float* conv_w, const float* conv_b,
                        const bf16_t* inT, const bf16_t* ssmT, const float* ssm_b,
                        const float* Amat, const float* Bv, const float* Cv,
                        const float* Dv, const bf16_t* outT, const float* out_b,
                        const float* resid, float* outC, bf16_t* outCbf,
                        float* xz, float* u, bf16_t* ubf, float* xssm,
                        float* cbuf, bf16_t* ybf, const bf16_t* zg)
{
    const int Mr = 16 * 2048, T = 2048, D = 1024, S = 32;
    dim3 blk(256);
    // in-proj: xz = X @ in_w + in_b      (N = 2D)
    conv_gemm_bf16<<<dim3(Mr / 256, (2 * D) / 128), blk, 0, stream>>>(
        Xbf, inT, in_b, xz, nullptr, nullptr, zg,
        Mr, 2 * D, D, T, 1, 0, 2 * D, 0, 0, 1.0f);
    // u = silu(dwconv_causal(xs))
    dwconv_silu<<<Mr * (D / 256), 256, 0, stream>>>(xz, conv_w, conv_b, u, ubf, T, D, 2 * D);
    // xssm = u @ ssm_w + ssm_b           (N = 32)
    conv_gemm_bf16<<<dim3(Mr / 256, 1), blk, 0, stream>>>(
        ubf, ssmT, ssm_b, xssm, nullptr, nullptr, zg,
        Mr, S, D, T, 1, 0, S, 0, 0, 1.0f);
    // sequential state scan -> c[b,t]
    scan_kernel<<<16, 32, 0, stream>>>(xssm, Amat, Bv, Cv, cbuf, T);
    // y = (c + Dv) * u * silu(gate)
    mamba_y<<<Mr * (D / 256), 256, 0, stream>>>(cbuf, Dv, u, xz, ybf, T, D, 2 * D);
    // out = y @ out_w + out_b + resid  (also emit bf16 shadow)
    conv_gemm_bf16<<<dim3(Mr / 256, D / 128), blk, 0, stream>>>(
        ybf, outT, out_b, outC, outCbf, resid, zg,
        Mr, D, D, T, 1, 0, D, 0, 0, 1.0f);
}

// ---------------------------------------------------------------------------
// kernel_launch
// ---------------------------------------------------------------------------
extern "C" void kernel_launch(void* const* d_in, const int* in_sizes, int n_in,
                              void* d_out, int out_size, void* d_ws, size_t ws_size,
                              hipStream_t stream) {
    (void)in_sizes; (void)n_in; (void)out_size; (void)ws_size;
    const int T = 2048, D = 1024, S = 32;
    const int Mr = 16 * T;                       // 32768 token rows
    const float bnscale = 1.0f / sqrtf(1.0f + 1e-5f);

    const float* x = (const float*)d_in[0];
    auto P = [&](int i) { return (const float*)d_in[i]; };
    const float* c1_w = P(25); const float* c1_b = P(26);
    const float* c4_w = P(27); const float* c5_w = P(28);
    const float* mem_p = P(29); const float* mem_s = P(30);
    const float* rp_w = P(31); const float* rp_b = P(32);

    // ---- workspace carve-out -------------------------------------------
    char* wp = (char*)d_ws;
    auto alloc = [&](size_t bytes) -> char* {
        char* r = wp; wp += (bytes + 255) & ~(size_t)255; return r;
    };
    bf16_t* zg    = (bf16_t*)alloc((size_t)4096 * 2);   // zero guard (>= Kd)
    bf16_t* bfX   = (bf16_t*)alloc((size_t)Mr * D * 2);
    float*  xz    = (float*) alloc((size_t)Mr * 2 * D * 4);
    float*  u     = (float*) alloc((size_t)Mr * D * 4);
    bf16_t* ubf   = (bf16_t*)alloc((size_t)Mr * D * 2);
    float*  xssm  = (float*) alloc((size_t)Mr * S * 4);
    float*  cbuf  = (float*) alloc((size_t)Mr * 4);
    bf16_t* ybf   = (bf16_t*)alloc((size_t)Mr * D * 2);
    float*  xm1   = (float*) alloc((size_t)Mr * D * 4);
    float*  xm2   = (float*) alloc((size_t)Mr * D * 4);
    bf16_t* ocbf  = (bf16_t*)alloc((size_t)Mr * D * 2);
    bf16_t* outbf = (bf16_t*)alloc((size_t)Mr * D * 2);
    bf16_t* in1t  = (bf16_t*)alloc((size_t)2 * D * D * 2);
    bf16_t* ssm1t = (bf16_t*)alloc((size_t)S * D * 2);
    bf16_t* out1t = (bf16_t*)alloc((size_t)D * D * 2);
    bf16_t* in2t  = (bf16_t*)alloc((size_t)2 * D * D * 2);
    bf16_t* ssm2t = (bf16_t*)alloc((size_t)S * D * 2);
    bf16_t* out2t = (bf16_t*)alloc((size_t)D * D * 2);
    bf16_t* c1t   = (bf16_t*)alloc((size_t)3 * 512 * D * 2);
    bf16_t* c4t   = (bf16_t*)alloc((size_t)512 * D * 2);
    bf16_t* c5t   = (bf16_t*)alloc((size_t)3 * D * D * 2);
    bf16_t* rpt   = (bf16_t*)alloc((size_t)D * D * 2);

    float* outP   = (float*)d_out;                      // (B,T,D) fp32
    float* anomP  = outP + (size_t)Mr * D;              // 16 floats
    float* reconP = anomP + 16;                         // (B,T,D) fp32

    auto cdiv = [](int a, int b) { return (a + b - 1) / b; };

    // ---- prep: zero guard, bf16 conversions, weight transposes ----------
    zero_bf16<<<cdiv(4096, 256), 256, 0, stream>>>(zg, 4096);
    cvt_f32_to_bf16<<<cdiv(Mr * D, 256), 256, 0, stream>>>(x, bfX, Mr * D);
    transpose_w<<<cdiv(D * 2 * D, 256), 256, 0, stream>>>(P(1),  in1t,  D, 2 * D);
    transpose_w<<<cdiv(D * S,     256), 256, 0, stream>>>(P(5),  ssm1t, D, S);
    transpose_w<<<cdiv(D * D,     256), 256, 0, stream>>>(P(11), out1t, D, D);
    transpose_w<<<cdiv(D * 2 * D, 256), 256, 0, stream>>>(P(13), in2t,  D, 2 * D);
    transpose_w<<<cdiv(D * S,     256), 256, 0, stream>>>(P(17), ssm2t, D, S);
    transpose_w<<<cdiv(D * D,     256), 256, 0, stream>>>(P(23), out2t, D, D);
    transpose_w<<<cdiv(D * D,     256), 256, 0, stream>>>(rp_w,  rpt,   D, D);
    prep_conv_w<<<cdiv(512 * D * 3, 256), 256, 0, stream>>>(c1_w, c1t, 512, D, 3);
    prep_conv_w<<<cdiv(512 * D * 1, 256), 256, 0, stream>>>(c4_w, c4t, 512, D, 1);
    prep_conv_w<<<cdiv(D * D * 3,   256), 256, 0, stream>>>(c5_w, c5t, D,   D, 3);

    // ---- Mamba block 1: xm1 = mamba(x),  bfX <- bf16(xm1) ---------------
    mamba_block(stream, bfX, P(2), P(3), P(4), in1t, ssm1t, P(6),
                P(7), P(8), P(9), P(10), out1t, P(12),
                x, xm1, bfX, xz, u, ubf, xssm, cbuf, ybf, zg);
    // ---- Mamba block 2: xm2 = mamba(xm1), bfX <- bf16(xm2) --------------
    mamba_block(stream, bfX, P(14), P(15), P(16), in2t, ssm2t, P(18),
                P(19), P(20), P(21), P(22), out2t, P(24),
                xm1, xm2, bfX, xz, u, ubf, xssm, cbuf, ybf, zg);

    // ---- conv stack ------------------------------------------------------
    dim3 blk(256);
    // o1 = bn(relu(conv3(o))) -> ocbf cols [0,512)
    conv_gemm_bf16<<<dim3(Mr / 256, 512 / 128), blk, 0, stream>>>(
        bfX, c1t, c1_b, nullptr, ocbf, nullptr, zg,
        Mr, 512, D, T, 3, -1, D, 0, 1, bnscale);
    // o4 = relu(conv1(o)) -> ocbf cols [512,1024)
    conv_gemm_bf16<<<dim3(Mr / 256, 512 / 128), blk, 0, stream>>>(
        bfX, c4t, nullptr, nullptr, ocbf, nullptr, zg,
        Mr, 512, D, T, 1, 0, D, 512, 1, 1.0f);
    // out = bn(relu(conv3(oc))) + residual -> d_out (fp32) + bf16 shadow
    conv_gemm_bf16<<<dim3(Mr / 256, D / 128), blk, 0, stream>>>(
        ocbf, c5t, nullptr, outP, outbf, xm2, zg,
        Mr, D, D, T, 3, -1, D, 0, 1, bnscale);

    // ---- anomaly head ----------------------------------------------------
    anomaly_kernel<<<16, 256, 0, stream>>>(outP, mem_p, mem_s, anomP, T, D);

    // ---- recon = out @ rp_w + rp_b --------------------------------------
    conv_gemm_bf16<<<dim3(Mr / 256, D / 128), blk, 0, stream>>>(
        outbf, rpt, rp_b, reconP, nullptr, nullptr, zg,
        Mr, D, D, T, 1, 0, D, 0, 0, 1.0f);
}
